// GCNdp4_31044023616332
// MI455X (gfx1250) — matrix operand order repaired and checked by reading, not compiled, not run on hardware
//
#include <hip/hip_runtime.h>

// ---------------- problem constants ----------------
constexpr int B_SZ  = 64;
constexpr int N0    = 491;
constexpr int NTOT  = B_SZ * N0;      // 31424
constexpr int E0    = N0 * 32;        // 15712 edges per graph
constexpr int F_IN  = 128;

// ---------------- types ----------------
typedef __attribute__((ext_vector_type(16))) __bf16 v16bf;
typedef __attribute__((ext_vector_type(8)))  __bf16 bf16x8;
typedef __attribute__((ext_vector_type(8)))  float  v8f;

// ---------------- helpers ----------------
static __device__ inline unsigned short f32_to_bf16(float f) {
  unsigned u = __float_as_uint(f);
  unsigned r = 0x7FFFu + ((u >> 16) & 1u);   // round-to-nearest-even
  return (unsigned short)((u + r) >> 16);
}
static __device__ inline unsigned enc_ord(float f) {  // order-preserving f32->u32
  unsigned u = __float_as_uint(f);
  return (u & 0x80000000u) ? ~u : (u | 0x80000000u);
}
static __device__ inline float dec_ord(unsigned u) {
  unsigned v = (u & 0x80000000u) ? (u & 0x7FFFFFFFu) : ~u;
  return __uint_as_float(v);
}
static __device__ inline float leaky(float v) { return v >= 0.f ? v : 0.2f * v; }

// ---------------- weight repack: [Cin,C] x4 (q,k,v,skip) -> WMMA B-layout bf16 ----------------
// packed idx = ((kt*colTiles + ct)*32 + lane)*16 + e
// lane<16: col=ct*16+lane, K = kt*32 + e ; lane>=16: col=ct*16+lane-16, K = kt*32 + 16 + e
__global__ void repack_wcat(const float* __restrict__ wq, const float* __restrict__ bq,
                            const float* __restrict__ wk, const float* __restrict__ bk,
                            const float* __restrict__ wv, const float* __restrict__ bv,
                            const float* __restrict__ ws, const float* __restrict__ bs,
                            int Cin, int C, unsigned short* __restrict__ packed,
                            float* __restrict__ bcat) {
  int Cout = 4 * C;
  int idx = blockIdx.x * blockDim.x + threadIdx.x;
  if (idx < Cout) {
    int g = idx / C, c = idx % C;
    const float* bb = (g == 0) ? bq : (g == 1) ? bk : (g == 2) ? bv : bs;
    bcat[idx] = bb[c];
  }
  int total = Cin * Cout;
  if (idx >= total) return;
  int e    = idx & 15;
  int lane = (idx >> 4) & 31;
  int rem  = idx >> 9;
  int colTiles = Cout >> 4;
  int ct = rem % colTiles;
  int kt = rem / colTiles;
  int col = (ct << 4) + (lane & 15);
  int K   = (kt << 5) + ((lane < 16) ? 0 : 16) + e;
  int g = col / C, c = col % C;
  const float* W = (g == 0) ? wq : (g == 1) ? wk : (g == 2) ? wv : ws;
  packed[idx] = f32_to_bf16(W[(size_t)K * C + c]);
}

// ---------------- build d1/d2 as bf16 [NTOT, 128] ----------------
__global__ void build_d(const float* __restrict__ x, const float* __restrict__ vk,
                        const float* __restrict__ Ww, const float* __restrict__ Wb,
                        const float* __restrict__ pos, int branch,
                        unsigned short* __restrict__ Db) {
  int idx = blockIdx.x * blockDim.x + threadIdx.x;     // NTOT*128
  if (idx >= NTOT * F_IN) return;
  int f = idx & (F_IN - 1);
  int n = idx >> 7;
  int b = n / N0;
  int i = n - b * N0;
  float v;
  if (i < N0 - 1) {
    v = x[(((size_t)b * 2 + branch) * (N0 - 1) + i) * F_IN + f];
  } else {
    float s = Wb[f];
    for (int t = 0; t < 64; ++t) s += vk[b * 64 + t] * Ww[t * F_IN + f];
    v = s;
  }
  Db[idx] = f32_to_bf16(v + pos[f]);
}

// ---------------- WMMA GEMM: Y[N,Cout] = X_bf16[N,Cin] @ Wpacked + bias ----------------
__global__ __launch_bounds__(32) void gemm_bf16_wmma(const unsigned short* __restrict__ Xu,
                                                     const unsigned short* __restrict__ Wu,
                                                     const float* __restrict__ bias,
                                                     float* __restrict__ Y,
                                                     int Cin, int Cout) {
  const __bf16* X  = (const __bf16*)Xu;
  const __bf16* Wp = (const __bf16*)Wu;
  int lane = threadIdx.x;
  int colTiles = Cout >> 4;
  int ct = blockIdx.x % colTiles;
  int rt = blockIdx.x / colTiles;
  int m0 = rt << 4;

  int row = m0 + (lane & 15);
  int c0  = (lane < 16) ? 0 : 16;
  const __bf16* xrow = X + (size_t)row * Cin;

  v8f acc = {};
  int kTiles = Cin >> 5;
  for (int kt = 0; kt < kTiles; ++kt) {
    int kb = kt << 5;
    union { v16bf v; bf16x8 h[2]; } A;
    A.h[0] = *(const bf16x8*)(xrow + kb + c0);
    A.h[1] = *(const bf16x8*)(xrow + kb + c0 + 8);
    v16bf Bm = *(const v16bf*)(Wp + (((size_t)kt * colTiles + ct) * 32 + lane) * 16);
    acc = __builtin_amdgcn_wmma_f32_16x16x32_bf16(false, A.v, false, Bm,
                                                  (short)0, acc, false, false);
  }
  int col = (ct << 4) + (lane & 15);
  float bv = bias[col];
  int rbase = m0 + ((lane < 16) ? 0 : 8);
  for (int i = 0; i < 8; ++i)
    Y[(size_t)(rbase + i) * Cout + col] = acc[i] + bv;
}

// ---------------- attention init: out = skip proj, m=-inf, den=0 ----------------
__global__ void init_attn(const float* __restrict__ Y4, int logC,
                          float* __restrict__ out, unsigned* __restrict__ m,
                          float* __restrict__ den) {
  int C = 1 << logC;
  int idx = blockIdx.x * blockDim.x + threadIdx.x;   // NTOT*C
  if (idx >= NTOT * C) return;
  int n = idx >> logC;
  int c = idx & (C - 1);
  out[idx] = Y4[(size_t)n * 4 * C + 3 * C + c];
  if (c == 0) { m[n] = 0x007FFFFFu; den[n] = 0.f; }
}

// ---------------- edge pass 1: score + segment max ----------------
__global__ void edge_score(const float* __restrict__ Y4, int C,
                           const int* __restrict__ ei, unsigned* __restrict__ m,
                           float* __restrict__ score) {
  int e0 = blockIdx.x * blockDim.x + threadIdx.x;
  if (e0 >= E0) return;
  int b = blockIdx.y, off = b * N0;
  int s = ei[e0] + off;
  int d = ei[E0 + e0] + off;
  const float* q = Y4 + (size_t)d * 4 * C;       // q block
  const float* k = Y4 + (size_t)s * 4 * C + C;   // k block
  float acc = 0.f;
  for (int c = 0; c < C; c += 4) {
    float4 qa = *(const float4*)(q + c);
    float4 ka = *(const float4*)(k + c);
    acc += qa.x * ka.x + qa.y * ka.y + qa.z * ka.z + qa.w * ka.w;
  }
  acc *= rsqrtf((float)C);
  score[(size_t)b * E0 + e0] = acc;
  atomicMax(m + d, enc_ord(acc));
}

// ---------------- edge pass 2: e = exp(score - m[dst]); den += e ----------------
__global__ void edge_exp(const int* __restrict__ ei, const unsigned* __restrict__ m,
                         float* __restrict__ score, float* __restrict__ den) {
  int e0 = blockIdx.x * blockDim.x + threadIdx.x;
  if (e0 >= E0) return;
  int b = blockIdx.y, off = b * N0;
  int d = ei[E0 + e0] + off;
  size_t g = (size_t)b * E0 + e0;
  float e = __expf(score[g] - dec_ord(m[d]));
  score[g] = e;
  atomicAdd(den + d, e);
}

// ---------------- edge pass 3: out[dst] += (e/den[dst]) * v[src] ----------------
__global__ void edge_agg(const float* __restrict__ Y4, int C, int chunkShift,
                         const int* __restrict__ ei, const float* __restrict__ ee,
                         const float* __restrict__ den, float* __restrict__ out) {
  int t = blockIdx.x * blockDim.x + threadIdx.x;     // E0 * (C/4)
  int chunks = C >> 2;
  if (t >= E0 * chunks) return;
  int cg = t & (chunks - 1);
  int e0 = t >> chunkShift;
  int b = blockIdx.y, off = b * N0;
  int s = ei[e0] + off;
  int d = ei[E0 + e0] + off;
  size_t g = (size_t)b * E0 + e0;
  float w = ee[g] / den[d];
  float4 v = *(const float4*)(Y4 + (size_t)s * 4 * C + 2 * C + cg * 4);
  float* o = out + (size_t)d * C + cg * 4;
  atomicAdd(o + 0, w * v.x);
  atomicAdd(o + 1, w * v.y);
  atomicAdd(o + 2, w * v.z);
  atomicAdd(o + 3, w * v.w);
}

// ---------------- leaky + bf16 convert ----------------
__global__ void leaky_bf16(const float* __restrict__ in, unsigned short* __restrict__ outb,
                           int count) {
  int idx = blockIdx.x * blockDim.x + threadIdx.x;
  if (idx >= count) return;
  outb[idx] = f32_to_bf16(leaky(in[idx]));
}

// ---------------- fc1: [64, 31424] @ [31424,100] + b, leaky ----------------
__global__ __launch_bounds__(256) void fc1_kernel(const float* __restrict__ h1,
                                                  const float* __restrict__ h2,
                                                  const float* __restrict__ W,
                                                  const float* __restrict__ bias,
                                                  float* __restrict__ hmid) {
  __shared__ float red[256];
  int blk = blockIdx.x;            // b*100 + j
  int b = blk / 100, j = blk - b * 100;
  const int half = N0 * 32;        // 15712
  const int K = 2 * half;
  float acc = 0.f;
  for (int k = threadIdx.x; k < K; k += 256) {
    float xv = (k < half) ? h1[(size_t)b * half + k] : h2[(size_t)b * half + (k - half)];
    acc += xv * W[(size_t)k * 100 + j];
  }
  red[threadIdx.x] = acc;
  __syncthreads();
  for (int s = 128; s > 0; s >>= 1) {
    if (threadIdx.x < s) red[threadIdx.x] += red[threadIdx.x + s];
    __syncthreads();
  }
  if (threadIdx.x == 0) hmid[b * 100 + j] = leaky(red[0] + bias[j]);
}

// ---------------- fc2: [64,100] @ [100,1] + b ----------------
__global__ void fc2_kernel(const float* __restrict__ hmid, const float* __restrict__ w,
                           const float* __restrict__ b2, float* __restrict__ out) {
  int b = blockIdx.x * blockDim.x + threadIdx.x;
  if (b >= B_SZ) return;
  float acc = b2[0];
  for (int j = 0; j < 100; ++j) acc += hmid[b * 100 + j] * w[j];
  out[b] = acc;
}

// =====================================================================
extern "C" void kernel_launch(void* const* d_in, const int* in_sizes, int n_in,
                              void* d_out, int out_size, void* d_ws, size_t ws_size,
                              hipStream_t stream) {
  // dict order: x, vk, edge_index, pos_enc, W1{w,b}, W2{w,b},
  //             t11{q,k,v,skip}{w,b}, t12{...}, t21{...}, t22{...}, fc1{w,b}, fc2{w,b}
  const float* x   = (const float*)d_in[0];
  const float* vk  = (const float*)d_in[1];
  const int*   ei  = (const int*)  d_in[2];
  const float* pos = (const float*)d_in[3];
  const float* W1w = (const float*)d_in[4];  const float* W1b = (const float*)d_in[5];
  const float* W2w = (const float*)d_in[6];  const float* W2b = (const float*)d_in[7];
  const float* t11[8]; for (int i = 0; i < 8; ++i) t11[i] = (const float*)d_in[8  + i];
  const float* t12[8]; for (int i = 0; i < 8; ++i) t12[i] = (const float*)d_in[16 + i];
  const float* t21[8]; for (int i = 0; i < 8; ++i) t21[i] = (const float*)d_in[24 + i];
  const float* t22[8]; for (int i = 0; i < 8; ++i) t22[i] = (const float*)d_in[32 + i];
  const float* fc1w = (const float*)d_in[40]; const float* fc1b = (const float*)d_in[41];
  const float* fc2w = (const float*)d_in[42]; const float* fc2b = (const float*)d_in[43];
  float* out = (float*)d_out;

  // ---- workspace carve (256B aligned) ----
  size_t off = 0;
  auto carve = [&](size_t bytes) -> char* {
    char* p = (char*)d_ws + off;
    off += (bytes + 255) & ~(size_t)255;
    return p;
  };
  unsigned short* pk11 = (unsigned short*)carve((size_t)128 * 256 * 2);
  unsigned short* pk12 = (unsigned short*)carve((size_t)64 * 128 * 2);
  unsigned short* pk21 = (unsigned short*)carve((size_t)128 * 256 * 2);
  unsigned short* pk22 = (unsigned short*)carve((size_t)64 * 128 * 2);
  float* bc11 = (float*)carve(256 * 4);
  float* bc12 = (float*)carve(128 * 4);
  float* bc21 = (float*)carve(256 * 4);
  float* bc22 = (float*)carve(128 * 4);
  unsigned short* db   = (unsigned short*)carve((size_t)NTOT * 128 * 2);
  float* Y4    = (float*)carve((size_t)NTOT * 256 * 4);
  float* out1  = (float*)carve((size_t)NTOT * 64 * 4);
  unsigned short* x2b = (unsigned short*)carve((size_t)NTOT * 64 * 2);
  float* h1    = (float*)carve((size_t)NTOT * 32 * 4);
  float* h2    = (float*)carve((size_t)NTOT * 32 * 4);
  float* score = (float*)carve((size_t)B_SZ * E0 * 4);
  unsigned* mbuf = (unsigned*)carve((size_t)NTOT * 4);
  float* den   = (float*)carve((size_t)NTOT * 4);
  float* hmid  = (float*)carve((size_t)B_SZ * 100 * 4);

  // ---- repack all tconv weights (tiny, once per launch) ----
  {
    int tot1 = 128 * 256, tot2 = 64 * 128;
    repack_wcat<<<(tot1 + 255) / 256, 256, 0, stream>>>(
        t11[0], t11[1], t11[2], t11[3], t11[4], t11[5], t11[6], t11[7], 128, 64, pk11, bc11);
    repack_wcat<<<(tot2 + 255) / 256, 256, 0, stream>>>(
        t12[0], t12[1], t12[2], t12[3], t12[4], t12[5], t12[6], t12[7], 64, 32, pk12, bc12);
    repack_wcat<<<(tot1 + 255) / 256, 256, 0, stream>>>(
        t21[0], t21[1], t21[2], t21[3], t21[4], t21[5], t21[6], t21[7], 128, 64, pk21, bc21);
    repack_wcat<<<(tot2 + 255) / 256, 256, 0, stream>>>(
        t22[0], t22[1], t22[2], t22[3], t22[4], t22[5], t22[6], t22[7], 64, 32, pk22, bc22);
  }

  const int rowTiles = NTOT / 16;                 // 1964
  for (int branch = 0; branch < 2; ++branch) {
    const float* Ww = branch ? W2w : W1w;
    const float* Wb = branch ? W2b : W1b;
    unsigned short* pkA = branch ? pk21 : pk11;
    float*          bcA = branch ? bc21 : bc11;
    unsigned short* pkB = branch ? pk22 : pk12;
    float*          bcB = branch ? bc22 : bc12;
    float* hout = branch ? h2 : h1;

    // d (bf16)
    build_d<<<(NTOT * 128 + 255) / 256, 256, 0, stream>>>(x, vk, Ww, Wb, pos, branch, db);

    // ---- tconv layer 1 (128 -> 64), fused q|k|v|skip GEMM (WMMA) ----
    gemm_bf16_wmma<<<rowTiles * (256 / 16), 32, 0, stream>>>(db, pkA, bcA, Y4, 128, 256);
    init_attn<<<(NTOT * 64 + 255) / 256, 256, 0, stream>>>(Y4, 6, out1, mbuf, den);
    {
      dim3 g1((E0 + 255) / 256, B_SZ);
      edge_score<<<g1, 256, 0, stream>>>(Y4, 64, ei, mbuf, score);
      edge_exp<<<g1, 256, 0, stream>>>(ei, mbuf, score, den);
      dim3 g2((E0 * 16 + 255) / 256, B_SZ);
      edge_agg<<<g2, 256, 0, stream>>>(Y4, 64, 4, ei, score, den, out1);
    }
    leaky_bf16<<<(NTOT * 64 + 255) / 256, 256, 0, stream>>>(out1, x2b, NTOT * 64);

    // ---- tconv layer 2 (64 -> 32) ----
    gemm_bf16_wmma<<<rowTiles * (128 / 16), 32, 0, stream>>>(x2b, pkB, bcB, Y4, 64, 128);
    init_attn<<<(NTOT * 32 + 255) / 256, 256, 0, stream>>>(Y4, 5, hout, mbuf, den);
    {
      dim3 g1((E0 + 255) / 256, B_SZ);
      edge_score<<<g1, 256, 0, stream>>>(Y4, 32, ei, mbuf, score);
      edge_exp<<<g1, 256, 0, stream>>>(ei, mbuf, score, den);
      dim3 g2((E0 * 8 + 255) / 256, B_SZ);
      edge_agg<<<g2, 256, 0, stream>>>(Y4, 32, 3, ei, score, den, hout);
    }
  }

  // ---- MLP head ----
  fc1_kernel<<<B_SZ * 100, 256, 0, stream>>>(h1, h2, fc1w, fc1b, hmid);
  fc2_kernel<<<1, 64, 0, stream>>>(hmid, fc2w, fc2b, out);

  (void)in_sizes; (void)n_in; (void)out_size; (void)ws_size;
}
